// LocalDictionary_26766236189394
// MI455X (gfx1250) — matrix operand order repaired and checked by reading, not compiled, not run on hardware
//
#include <hip/hip_runtime.h>

#define BATCH   2048
#define INSZ    512
#define HID     1024
#define NLAYERS 30
#define PEN     0.1f
#define ROWS    16          // batch rows per workgroup
#define XSTRIDE 1032        // bf16 elems per xbf row: 1024 + 8 pad -> 516 DW stride (bank-friendly)
#define XCHS    1040        // f32 elems per xch row: 1024 + 16 pad -> disjoint bank halves
#define NEWTON_ITERS 12

typedef __bf16 bf16_t;
typedef __attribute__((ext_vector_type(16))) __bf16 v16bf;
typedef __attribute__((ext_vector_type(8)))  float  v8f;

union FragU { uint4 u[2]; v16bf v; };

__device__ __forceinline__ bf16_t f2bf(float f) {
  unsigned u = __builtin_bit_cast(unsigned, f);
  u += 0x7fffu + ((u >> 16) & 1u);              // round-to-nearest-even
  unsigned short h = (unsigned short)(u >> 16);
  return __builtin_bit_cast(bf16_t, h);
}

__device__ __forceinline__ v8f v8zero() {
  v8f v;
#pragma unroll
  for (int i = 0; i < 8; ++i) v[i] = 0.0f;
  return v;
}

__device__ __forceinline__ float wave_sum(float x) {
#pragma unroll
  for (int m = 16; m >= 1; m >>= 1) x += __shfl_xor(x, m, 32);
  return x;
}

// A fragment, 16x32 bf16 tile from row-major storage (ld in elements).
// Per ISA layout: lane m=(lane&15), koff=(lane>>4)*8; chunk0 K=[k0+koff,+8), chunk1 K=[k0+16+koff,+8)
__device__ __forceinline__ v16bf load_afrag(const bf16_t* base, int ld, int m0, int k0, int lane) {
  const bf16_t* p = base + (size_t)(m0 + (lane & 15)) * ld + k0 + ((lane >> 4) << 3);
  FragU f;
  f.u[0] = *(const uint4*)(p);
  f.u[1] = *(const uint4*)(p + 16);
  return f.v;
}

// B fragment, 32x16 bf16 tile; operand stored as "rows of K" (i.e. B^T row-major).
// lane n=(lane&15), koff=(lane>>4)*8; chunk0 K=[k0+koff,+8), chunk1 K=[k0+16+koff,+8)
__device__ __forceinline__ v16bf load_bfrag(const bf16_t* base, int ld, int n0, int k0, int lane) {
  const bf16_t* p = base + (size_t)(n0 + (lane & 15)) * ld + k0 + ((lane >> 4) << 3);
  FragU f;
  f.u[0] = *(const uint4*)(p);
  f.u[1] = *(const uint4*)(p + 16);
  return f.v;
}

__device__ __forceinline__ v8f wmma_bf16(v16bf a, v16bf b, v8f c) {
  return __builtin_amdgcn_wmma_f32_16x16x32_bf16(false, a, false, b, (short)0, c, false, false);
}

// ---------------- prep kernels ----------------

// bf16 conversions: ybf [2048][512], Wbf [1024][512], Wtbf [512][1024] (transposed W)
__global__ void k_convert(const float* __restrict__ W, const float* __restrict__ y,
                          bf16_t* __restrict__ Wbf, bf16_t* __restrict__ Wtbf,
                          bf16_t* __restrict__ ybf) {
  int idx = blockIdx.x * 256 + threadIdx.x;
  if (idx < BATCH * INSZ) ybf[idx] = f2bf(y[idx]);
  if (idx < HID * INSZ) {
    float w = W[idx];
    Wbf[idx] = f2bf(w);
    int h = idx >> 9, i = idx & 511;
    Wtbf[i * HID + h] = f2bf(w);
  }
}

// row norms: ynorm[2048], wnorm[1024]; one wave per row
__global__ void k_norms(const float* __restrict__ y, const float* __restrict__ W,
                        float* __restrict__ ynorm, float* __restrict__ wnorm) {
  int wid  = (blockIdx.x * 256 + threadIdx.x) >> 5;
  int lane = threadIdx.x & 31;
  if (wid < BATCH) {
    const float* r = y + (size_t)wid * INSZ;
    float s = 0.f;
    for (int k = lane; k < INSZ; k += 32) { float v = r[k]; s = fmaf(v, v, s); }
    s = wave_sum(s);
    if (lane == 0) ynorm[wid] = s;
  } else if (wid < BATCH + HID) {
    int h = wid - BATCH;
    const float* r = W + (size_t)h * INSZ;
    float s = 0.f;
    for (int k = lane; k < INSZ; k += 32) { float v = r[k]; s = fmaf(v, v, s); }
    s = wave_sum(s);
    if (lane == 0) wnorm[h] = s;
  }
}

// Gram matrix: Gbf = bf16(W @ W^T), [1024][1024]; one 16x16 tile per wave
__global__ __launch_bounds__(256) void k_gram(const bf16_t* __restrict__ Wbf,
                                              bf16_t* __restrict__ Gbf) {
  int wid  = (blockIdx.x * 256 + threadIdx.x) >> 5;
  int lane = threadIdx.x & 31;
  int m0 = (wid >> 6) * 16, n0 = (wid & 63) * 16;
  v8f acc = v8zero();
  for (int ks = 0; ks < INSZ / 32; ++ks) {
    int k0 = ks * 32;
    v16bf a = load_afrag(Wbf, INSZ, m0, k0, lane);
    v16bf b = load_bfrag(Wbf, INSZ, n0, k0, lane);   // B[k][n] = W[n][k]
    acc = wmma_bf16(a, b, acc);
  }
  int rr = (lane >> 4) << 3, cc = lane & 15;
#pragma unroll
  for (int j = 0; j < 8; ++j)
    Gbf[(size_t)(m0 + j + rr) * HID + n0 + cc] = f2bf(acc[j]);
}

// c = PEN*(||y_m||^2 + ||W_j||^2) - (1+2*PEN)*(y @ W^T), [2048][1024] f32
__global__ __launch_bounds__(256) void k_cmat(const bf16_t* __restrict__ ybf,
                                              const bf16_t* __restrict__ Wbf,
                                              const float* __restrict__ ynorm,
                                              const float* __restrict__ wnorm,
                                              float* __restrict__ cbuf) {
  int wid  = (blockIdx.x * 256 + threadIdx.x) >> 5;
  int lane = threadIdx.x & 31;
  int m0 = (wid >> 6) * 16, n0 = (wid & 63) * 16;
  v8f acc = v8zero();
  for (int ks = 0; ks < INSZ / 32; ++ks) {
    int k0 = ks * 32;
    v16bf a = load_afrag(ybf, INSZ, m0, k0, lane);
    v16bf b = load_bfrag(Wbf, INSZ, n0, k0, lane);
    acc = wmma_bf16(a, b, acc);
  }
  int rr = (lane >> 4) << 3, cc = lane & 15;
#pragma unroll
  for (int j = 0; j < 8; ++j) {
    int row = m0 + j + rr, col = n0 + cc;
    cbuf[(size_t)row * HID + col] =
        PEN * (ynorm[row] + wnorm[col]) - (1.0f + 2.0f * PEN) * acc[j];
  }
}

// ---------------- persistent LISTA kernel ----------------
// 128 blocks x 256 threads (8 wave32). Block owns 16 batch rows for all 30 layers.
__global__ __launch_bounds__(256, 1)
void k_lista(const bf16_t* __restrict__ Gbf, const bf16_t* __restrict__ Wtbf,
             const float* __restrict__ cbuf, const float* __restrict__ stepp,
             float* __restrict__ out) {
  extern __shared__ char smem[];
  bf16_t* xbf = (bf16_t*)smem;                          // [ROWS][XSTRIDE] bf16 (GEMM A operand)
  float*  xch = (float*)(smem + ROWS * XSTRIDE * 2);    // [ROWS][XCHS] f32  (grad exchange)

  const int tid  = threadIdx.x;
  const int lane = tid & 31;
  const int wv   = tid >> 5;
  const int m0   = blockIdx.x * ROWS;
  const float step = stepp[0];

  for (int i = tid; i < ROWS * XSTRIDE; i += 256) xbf[i] = f2bf(0.0f);

  // per-wave activation state: rows 2*wv, 2*wv+1; lane holds cols {lane + 32k}
  float xtmp[2][32], xold[2][32];
#pragma unroll
  for (int r = 0; r < 2; ++r)
#pragma unroll
    for (int k = 0; k < 32; ++k) { xtmp[r][k] = 0.f; xold[r][k] = 0.f; }

  __syncthreads();

  const int n0 = wv * 128;                 // wave's 128-col slice of HID
  const int rr = (lane >> 4) << 3;
  const int cc = lane & 15;

  for (int l = 0; l < NLAYERS; ++l) {
    // ---- GEMM: p = x_tmp @ G  (G symmetric -> B frags read from row-major G) ----
    v8f acc[8];
#pragma unroll
    for (int t = 0; t < 8; ++t) acc[t] = v8zero();

    for (int ks = 0; ks < HID / 32; ++ks) {
      const int k0 = ks * 32;
      v16bf a = load_afrag(xbf, XSTRIDE, 0, k0, lane);
      v16bf b[8];
#pragma unroll
      for (int t = 0; t < 8; ++t) b[t] = load_bfrag(Gbf, HID, n0 + t * 16, k0, lane);
#pragma unroll
      for (int t = 0; t < 8; ++t) acc[t] = wmma_bf16(a, b[t], acc[t]);
    }

#pragma unroll
    for (int t = 0; t < 8; ++t)
#pragma unroll
      for (int j = 0; j < 8; ++j)
        xch[(j + rr) * XCHS + n0 + t * 16 + cc] = acc[t][j];

    __syncthreads();

    // ---- activation: sparsemax via Newton on sum(relu(v - tau)) = 1 ----
    const float mom  = (float)l / ((float)l + 3.0f);
    const bool  last = (l == NLAYERS - 1);

#pragma unroll
    for (int r = 0; r < 2; ++r) {
      const int row = 2 * wv + r;
      const float* crow = cbuf + (size_t)(m0 + row) * HID;
      float cand[32];
      float ssum = 0.f;
#pragma unroll
      for (int k = 0; k < 32; ++k) {
        float g  = xch[row * XCHS + lane + 32 * k] + crow[lane + 32 * k]; // grad = p + c
        float cd = xtmp[r][k] - step * g;
        cand[k] = cd;
        ssum += cd;
      }
      float tau = (wave_sum(ssum) - 1.0f) * (1.0f / (float)HID);  // tau0 <= tau*
      for (int it = 0; it < NEWTON_ITERS; ++it) {                 // monotone, exact on convergence
        float s = 0.f, c = 0.f;
#pragma unroll
        for (int k = 0; k < 32; ++k) {
          bool gt = cand[k] > tau;
          s += gt ? cand[k] : 0.f;
          c += gt ? 1.f : 0.f;
        }
        s = wave_sum(s);
        c = wave_sum(c);
        tau = (s - 1.0f) / c;                                     // c >= 1 guaranteed
      }
#pragma unroll
      for (int k = 0; k < 32; ++k) {
        float xn = fmaxf(cand[k] - tau, 0.f);
        float xt = xn + mom * (xn - xold[r][k]);
        xold[r][k] = xn;
        xtmp[r][k] = xt;
        xbf[row * XSTRIDE + lane + 32 * k] = f2bf(last ? xn : xt); // decode uses x_new
      }
    }
    __syncthreads();
  }

  // ---- decode: out = x_new @ W ; rows-of-K operand = Wtbf [INSZ][HID] ----
  {
    const int n0d = wv * 64;    // 512 cols over 8 waves
    v8f acc[4];
#pragma unroll
    for (int t = 0; t < 4; ++t) acc[t] = v8zero();
    for (int ks = 0; ks < HID / 32; ++ks) {
      const int k0 = ks * 32;
      v16bf a = load_afrag(xbf, XSTRIDE, 0, k0, lane);
#pragma unroll
      for (int t = 0; t < 4; ++t) {
        v16bf b = load_bfrag(Wtbf, HID, n0d + t * 16, k0, lane);
        acc[t] = wmma_bf16(a, b, acc[t]);
      }
    }
#pragma unroll
    for (int t = 0; t < 4; ++t)
#pragma unroll
      for (int j = 0; j < 8; ++j)
        out[(size_t)(m0 + j + rr) * INSZ + n0d + t * 16 + cc] = acc[t][j];
  }
}

// ---------------- launch ----------------
extern "C" void kernel_launch(void* const* d_in, const int* in_sizes, int n_in,
                              void* d_out, int out_size, void* d_ws, size_t ws_size,
                              hipStream_t stream) {
  (void)in_sizes; (void)n_in; (void)out_size; (void)ws_size;
  const float* y    = (const float*)d_in[0];   // [2048,512]
  const float* W    = (const float*)d_in[1];   // [1024,512]
  const float* step = (const float*)d_in[2];   // scalar
  float* out = (float*)d_out;                  // [2048,512]

  char* ws = (char*)d_ws;
  bf16_t* Wbf   = (bf16_t*)(ws + 0);           // 1,048,576
  bf16_t* ybf   = (bf16_t*)(ws + 1048576);     // 2,097,152
  bf16_t* Wtbf  = (bf16_t*)(ws + 3145728);     // 1,048,576
  bf16_t* Gbf   = (bf16_t*)(ws + 4194304);     // 2,097,152
  float*  cbuf  = (float*) (ws + 6291456);     // 8,388,608
  float*  ynorm = (float*) (ws + 14680064);    // 8,192
  float*  wnorm = (float*) (ws + 14688256);    // 4,096  (total ~14.7 MB)

  k_convert<<<(BATCH * INSZ) / 256, 256, 0, stream>>>(W, y, Wbf, Wtbf, ybf);
  k_norms<<<(BATCH + HID) * 32 / 256, 256, 0, stream>>>(y, W, ynorm, wnorm);
  k_gram<<<(HID / 16) * (HID / 16) / 8, 256, 0, stream>>>(Wbf, Gbf);
  k_cmat<<<(BATCH / 16) * (HID / 16) / 8, 256, 0, stream>>>(ybf, Wbf, ynorm, wnorm, cbuf);

  const int smem_bytes = ROWS * XSTRIDE * 2 + ROWS * XCHS * 4;  // 33,024 + 66,560 = 99,584
  k_lista<<<BATCH / ROWS, 256, smem_bytes, stream>>>(Gbf, Wtbf, cbuf, step, out);
}
